// Transformer_28398323761700
// MI455X (gfx1250) — matrix operand order repaired and checked by reading, not compiled, not run on hardware
//
#include <hip/hip_runtime.h>
#include <hip/hip_bf16.h>

// ---------------------------------------------------------------------------
// Types for CDNA5 WMMA
// ---------------------------------------------------------------------------
typedef __attribute__((ext_vector_type(16))) __bf16          v16bf;
typedef __attribute__((ext_vector_type(16))) _Float16        v16h;
typedef __attribute__((ext_vector_type(8)))  float           v8f;
typedef __attribute__((ext_vector_type(8)))  unsigned short  ush8;

union BFrag { ush8 h[2]; v16bf v; };   // 16 bf16 = 8 VGPRs
union HFrag { ush8 h[2]; v16h  v; };   // 16 f16  = 8 VGPRs

__device__ __forceinline__ v8f vzero8() {
    v8f z = {0.f, 0.f, 0.f, 0.f, 0.f, 0.f, 0.f, 0.f};
    return z;
}

__device__ __forceinline__ unsigned short f2bf(float f) {
    unsigned int u = __float_as_uint(f);
    unsigned int r = u + 0x7fffu + ((u >> 16) & 1u);   // RNE
    return (unsigned short)(r >> 16);
}
__device__ __forceinline__ float bf2f(unsigned short b) {
    return __uint_as_float(((unsigned int)b) << 16);
}
__device__ __forceinline__ unsigned short f2h_bits(float x) {
    union { _Float16 f; unsigned short u; } c;
    c.f = (_Float16)x;
    return c.u;
}

// ---------------------------------------------------------------------------
// DPP16 butterfly reductions within each 16-lane group (no LDS round-trips).
// quad_perm xor1 = 0xB1, quad_perm xor2 = 0x4E,
// row_half_mirror = 0x141, row_mirror = 0x140.
// All 16 lanes converge to the reduced value.
// ---------------------------------------------------------------------------
template <int CTRL>
__device__ __forceinline__ float dppmov(float x) {
    return __int_as_float(__builtin_amdgcn_update_dpp(
        0, __float_as_int(x), CTRL, 0xF, 0xF, true));
}
__device__ __forceinline__ float redmax16(float x) {
    x = fmaxf(x, dppmov<0xB1>(x));
    x = fmaxf(x, dppmov<0x4E>(x));
    x = fmaxf(x, dppmov<0x141>(x));
    x = fmaxf(x, dppmov<0x140>(x));
    return x;
}
__device__ __forceinline__ float redsum16(float x) {
    x += dppmov<0xB1>(x);
    x += dppmov<0x4E>(x);
    x += dppmov<0x141>(x);
    x += dppmov<0x140>(x);
    return x;
}

// ---------------------------------------------------------------------------
// Model constants
// ---------------------------------------------------------------------------
#define DIM    1024
#define MLP_D  4096
#define HEADS  16
#define DH     64
#define SEQ    2048
#define BATCH  2
#define ROWS   (BATCH * SEQ)      // 4096
#define QKV3   (3 * DIM)          // 3072

// ---------------------------------------------------------------------------
// fp32 [R][C] -> bf16 TRANSPOSED [C][R], LDS-tiled (coalesced both ways).
// blockIdx.z = depth slice (stride R*C).  R, C multiples of 32.
// ---------------------------------------------------------------------------
__global__ __launch_bounds__(256)
void transpose_f32_to_bf16(const float* __restrict__ in,
                           unsigned short* __restrict__ out, int R, int C) {
    __shared__ float tile[32][33];
    const size_t zoff = (size_t)blockIdx.z * R * C;
    const int tx = threadIdx.x & 31, ty = threadIdx.x >> 5;   // 32 x 8
    const int c0 = blockIdx.x * 32, r0 = blockIdx.y * 32;
    const float* src = in + zoff;
    unsigned short* dst = out + zoff;
#pragma unroll
    for (int i = 0; i < 32; i += 8)
        tile[ty + i][tx] = src[(size_t)(r0 + ty + i) * C + c0 + tx];
    __syncthreads();
#pragma unroll
    for (int i = 0; i < 32; i += 8)
        dst[(size_t)(c0 + ty + i) * R + r0 + tx] = f2bf(tile[tx][ty + i]);
}

// ---------------------------------------------------------------------------
// LayerNorm (fp32 residual in) -> bf16 out.  One block per row, 256 threads.
// ---------------------------------------------------------------------------
__global__ __launch_bounds__(256)
void layernorm_to_bf16(const float* __restrict__ x,
                       const float* __restrict__ gm,
                       const float* __restrict__ bt,
                       unsigned short* __restrict__ out) {
    __shared__ float red[16];
    const int row = blockIdx.x, t = threadIdx.x;
    const float4 v = *(const float4*)(x + (size_t)row * DIM + t * 4);
    float s = v.x + v.y + v.z + v.w;
    float q = v.x * v.x + v.y * v.y + v.z * v.z + v.w * v.w;
#pragma unroll
    for (int off = 16; off; off >>= 1) {
        s += __shfl_xor(s, off, 32);
        q += __shfl_xor(q, off, 32);
    }
    const int wave = t >> 5, lane = t & 31;
    if (lane == 0) { red[wave] = s; red[8 + wave] = q; }
    __syncthreads();
    if (t == 0) {
        float S = 0.f, Q = 0.f;
        for (int i = 0; i < 8; ++i) { S += red[i]; Q += red[8 + i]; }
        float mean = S * (1.f / DIM);
        float var  = Q * (1.f / DIM) - mean * mean;
        red[0] = mean;
        red[1] = rsqrtf(var + 1e-5f);
    }
    __syncthreads();
    const float mean = red[0], rstd = red[1];
    const int c = t * 4;
    const float4 gv = *(const float4*)(gm + c);
    const float4 bv = *(const float4*)(bt + c);
    unsigned short* op = out + (size_t)row * DIM + c;
    op[0] = f2bf((v.x - mean) * rstd * gv.x + bv.x);
    op[1] = f2bf((v.y - mean) * rstd * gv.y + bv.y);
    op[2] = f2bf((v.z - mean) * rstd * gv.z + bv.z);
    op[3] = f2bf((v.w - mean) * rstd * gv.w + bv.w);
}

// ---------------------------------------------------------------------------
// bf16 WMMA GEMM: C[M,Nc] = A[M,K] x Wt[Nc,K]^T  (+bias, +GELU, +residual)
// Wt is PRE-TRANSPOSED ([n][k]) so both tiles stage with contiguous b128.
// Block tile 128x128, BK=32, 256 threads (8 waves, 4x2), wave tile 32x64.
// Double-buffered LDS: one barrier per K-step, loads overlap WMMA.
// ---------------------------------------------------------------------------
#define BM 128
#define BN 128
#define BK 32
#define LDT (BK + 8)   // 40 halfwords: 16B-aligned rows, conflict-free b128

template <int BIAS, int GELU, int ADD, int OUTBF>
__global__ __launch_bounds__(256)
void gemm_bf16_wmma(const unsigned short* __restrict__ A,
                    const unsigned short* __restrict__ Wt,
                    const float* __restrict__ bias,
                    float* __restrict__ outf,
                    unsigned short* __restrict__ outb,
                    int M, int Ncols, int K) {
    __shared__ alignas(16) unsigned short As[2][BM][LDT];  // [m][k]
    __shared__ alignas(16) unsigned short Bs[2][BN][LDT];  // [n][k]

    const int t = threadIdx.x;
    const int lane = t & 31, wave = t >> 5;
    const int wm = wave >> 1, wn = wave & 1;             // 4x2 waves
    const int m0 = blockIdx.y * BM, n0 = blockIdx.x * BN;
    const int g = lane >> 4, l16 = lane & 15;

    v8f acc[2][4];
#pragma unroll
    for (int i = 0; i < 2; ++i)
#pragma unroll
        for (int j = 0; j < 4; ++j) acc[i][j] = vzero8();

    const int srow = t >> 1, skseg = (t & 1) * 16;       // 2 thr/row x 16

    const unsigned short* aptr = A  + (size_t)(m0 + srow) * K + skseg;
    const unsigned short* bptr = Wt + (size_t)(n0 + srow) * K + skseg;

    // prime buffer 0
    {
        ush8 a0 = *(const ush8*)aptr;
        ush8 a1 = *(const ush8*)(aptr + 8);
        ush8 b0 = *(const ush8*)bptr;
        ush8 b1 = *(const ush8*)(bptr + 8);
        *(ush8*)&As[0][srow][skseg]     = a0;
        *(ush8*)&As[0][srow][skseg + 8] = a1;
        *(ush8*)&Bs[0][srow][skseg]     = b0;
        *(ush8*)&Bs[0][srow][skseg + 8] = b1;
    }

    int buf = 0;
    for (int k0 = 0; k0 < K; k0 += BK) {
        __syncthreads();
        if (k0 + BK < K) {   // stage next tile into the other buffer
            const unsigned short* an = aptr + k0 + BK;
            const unsigned short* bn = bptr + k0 + BK;
            ush8 a0 = *(const ush8*)an;
            ush8 a1 = *(const ush8*)(an + 8);
            ush8 b0 = *(const ush8*)bn;
            ush8 b1 = *(const ush8*)(bn + 8);
            *(ush8*)&As[buf ^ 1][srow][skseg]     = a0;
            *(ush8*)&As[buf ^ 1][srow][skseg + 8] = a1;
            *(ush8*)&Bs[buf ^ 1][srow][skseg]     = b0;
            *(ush8*)&Bs[buf ^ 1][srow][skseg + 8] = b1;
        }

        BFrag a[2], b[4];
#pragma unroll
        for (int tm = 0; tm < 2; ++tm) {
            // A 16x32 bf16 layout: elems 0..7 -> K = g*8+t ; 8..15 -> K = 16+g*8+t
            const int m = wm * 32 + tm * 16 + l16;
            a[tm].h[0] = *(const ush8*)&As[buf][m][g * 8];
            a[tm].h[1] = *(const ush8*)&As[buf][m][16 + g * 8];
        }
#pragma unroll
        for (int tn = 0; tn < 4; ++tn) {
            // B 32x16 bf16 layout: lane N = l16, elems 0..7 -> K = g*8+t ; 8..15 -> K = 16+g*8+t
            const int n = wn * 64 + tn * 16 + l16;
            b[tn].h[0] = *(const ush8*)&Bs[buf][n][g * 8];
            b[tn].h[1] = *(const ush8*)&Bs[buf][n][16 + g * 8];
        }
#pragma unroll
        for (int tm = 0; tm < 2; ++tm)
#pragma unroll
            for (int tn = 0; tn < 4; ++tn)
                acc[tm][tn] = __builtin_amdgcn_wmma_f32_16x16x32_bf16(
                    false, a[tm].v, false, b[tn].v, (short)0, acc[tm][tn],
                    false, false);
        buf ^= 1;
    }

    // Epilogue: C layout -> M = v + 8*g, N = l16 within each 16x16 tile
#pragma unroll
    for (int tm = 0; tm < 2; ++tm)
#pragma unroll
        for (int tn = 0; tn < 4; ++tn) {
            const int col = n0 + wn * 64 + tn * 16 + l16;
            const float bv = BIAS ? bias[col] : 0.f;
#pragma unroll
            for (int v = 0; v < 8; ++v) {
                const int row = m0 + wm * 32 + tm * 16 + v + 8 * g;
                float c = acc[tm][tn][v] + bv;
                if (GELU) c = 0.5f * c * (1.f + erff(c * 0.70710678118f));
                const size_t idx = (size_t)row * Ncols + col;
                if (ADD) c += outf[idx];
                if (OUTBF) outb[idx] = f2bf(c);
                else       outf[idx] = c;
            }
        }
}

// ---------------------------------------------------------------------------
// Flash-style attention.  qkv: bf16 [ROWS, 3072] (q|k|v per head inside row).
// Grid (N/64, HEADS, B), 128 threads = 4 waves; wave owns 16 query rows.
// Online softmax of UNSCALED logits; reference divides softmax by SCALE
// before PV, so fold a x8 into the finalize.
// Row reductions are DPP16 butterflies (row M = v + 8*g lives in one
// 16-lane half of the C layout).
// ---------------------------------------------------------------------------
#define LDV 72   // 64 + 8 halfword pad

__global__ __launch_bounds__(128)
void attention_wmma(const unsigned short* __restrict__ qkv,
                    unsigned short* __restrict__ o) {
    __shared__ alignas(16) unsigned short Vt[DH][LDV];       // V^T [d][k], f16
    __shared__ alignas(16) unsigned short Pl[4][16][LDV];    // per-wave P, f16

    const int t = threadIdx.x, lane = t & 31, w = t >> 5;
    const int g = lane >> 4, l16 = lane & 15;
    const int qt = blockIdx.x, h = blockIdx.y, b = blockIdx.z;
    const size_t rowbase = (size_t)b * SEQ;
    const int qbase = qt * 64;

    // Q fragments for this wave's 16 rows (kept in regs for all k-tiles)
    BFrag qa[2];
    {
        const int qrow = qbase + w * 16 + l16;
        const unsigned short* qp = qkv + (rowbase + qrow) * QKV3 + h * DH;
#pragma unroll
        for (int s = 0; s < 2; ++s) {
            qa[s].h[0] = *(const ush8*)(qp + s * 32 + g * 8);
            qa[s].h[1] = *(const ush8*)(qp + s * 32 + 16 + g * 8);
        }
    }

    v8f Oacc[4];
#pragma unroll
    for (int i = 0; i < 4; ++i) Oacc[i] = vzero8();
    float m_row[8], l_row[8];
#pragma unroll
    for (int v = 0; v < 8; ++v) { m_row[v] = -1e30f; l_row[v] = 0.f; }

    const int vk = t >> 1, vdseg = (t & 1) * 32;   // V staging map

    for (int kt = 0; kt < SEQ / 64; ++kt) {
        const int kbase = kt * 64;
        __syncthreads();   // protect Vt vs previous iteration's reads
        {   // stage V tile transposed, bf16 -> f16
            const unsigned short* vp =
                qkv + (rowbase + kbase + vk) * QKV3 + 2 * DIM + h * DH + vdseg;
            ush8 x0 = *(const ush8*)vp;
            ush8 x1 = *(const ush8*)(vp + 8);
            ush8 x2 = *(const ush8*)(vp + 16);
            ush8 x3 = *(const ush8*)(vp + 24);
#pragma unroll
            for (int j = 0; j < 8; ++j) Vt[vdseg + j][vk]      = f2h_bits(bf2f(x0[j]));
#pragma unroll
            for (int j = 0; j < 8; ++j) Vt[vdseg + 8 + j][vk]  = f2h_bits(bf2f(x1[j]));
#pragma unroll
            for (int j = 0; j < 8; ++j) Vt[vdseg + 16 + j][vk] = f2h_bits(bf2f(x2[j]));
#pragma unroll
            for (int j = 0; j < 8; ++j) Vt[vdseg + 24 + j][vk] = f2h_bits(bf2f(x3[j]));
        }
        __syncthreads();

        // S = Q K^T : bf16 WMMA, B-frags straight from global (k contiguous)
        v8f S[4];
#pragma unroll
        for (int i = 0; i < 4; ++i) S[i] = vzero8();
#pragma unroll
        for (int tn = 0; tn < 4; ++tn) {
            const unsigned short* kp =
                qkv + (rowbase + kbase + tn * 16 + l16) * QKV3 + DIM + h * DH;
#pragma unroll
            for (int s = 0; s < 2; ++s) {
                BFrag kb;
                kb.h[0] = *(const ush8*)(kp + s * 32 + g * 8);
                kb.h[1] = *(const ush8*)(kp + s * 32 + 16 + g * 8);
                S[tn] = __builtin_amdgcn_wmma_f32_16x16x32_bf16(
                    false, qa[s].v, false, kb.v, (short)0, S[tn], false, false);
            }
        }

        // Online softmax, DPP16 reductions (uniform within each 16-lane half)
#pragma unroll
        for (int v = 0; v < 8; ++v) {
            float smax = fmaxf(fmaxf(S[0][v], S[1][v]), fmaxf(S[2][v], S[3][v]));
            smax = redmax16(smax);
            const float mnew  = fmaxf(m_row[v], smax);
            const float alpha = __expf(m_row[v] - mnew);
            float rs = 0.f;
#pragma unroll
            for (int tn = 0; tn < 4; ++tn) {
                const float p = __expf(S[tn][v] - mnew);
                S[tn][v] = p;
                rs += p;
            }
            rs = redsum16(rs);
            l_row[v] = l_row[v] * alpha + rs;
            m_row[v] = mnew;
#pragma unroll
            for (int td = 0; td < 4; ++td) Oacc[td][v] *= alpha;
        }

        // P -> LDS as f16 (wave-private region)
#pragma unroll
        for (int tn = 0; tn < 4; ++tn)
#pragma unroll
            for (int v = 0; v < 8; ++v)
                Pl[w][v + 8 * g][tn * 16 + l16] = f2h_bits(S[tn][v]);
        __syncthreads();

        // O += P V : f16 WMMA
#pragma unroll
        for (int s = 0; s < 2; ++s) {
            HFrag pa;
            pa.h[0] = *(const ush8*)&Pl[w][l16][s * 32 + g * 8];
            pa.h[1] = *(const ush8*)&Pl[w][l16][s * 32 + 16 + g * 8];
#pragma unroll
            for (int td = 0; td < 4; ++td) {
                HFrag vb;
                vb.h[0] = *(const ush8*)&Vt[td * 16 + l16][s * 32 + g * 8];
                vb.h[1] = *(const ush8*)&Vt[td * 16 + l16][s * 32 + 16 + g * 8];
                Oacc[td] = __builtin_amdgcn_wmma_f32_16x16x32_f16(
                    false, pa.v, false, vb.v, (short)0, Oacc[td], false, false);
            }
        }
    }

    // finalize: O = (Oacc / l) * (1/SCALE = 8)
#pragma unroll
    for (int v = 0; v < 8; ++v) {
        const float inv = 8.0f / l_row[v];
        const int qrow = qbase + w * 16 + v + 8 * g;
        unsigned short* op = o + (rowbase + qrow) * DIM + h * DH;
#pragma unroll
        for (int td = 0; td < 4; ++td)
            op[td * 16 + l16] = f2bf(Oacc[td][v] * inv);
    }
}

// ---------------------------------------------------------------------------
// Host orchestration
// ---------------------------------------------------------------------------
extern "C" void kernel_launch(void* const* d_in, const int* in_sizes, int n_in,
                              void* d_out, int out_size, void* d_ws, size_t ws_size,
                              hipStream_t stream) {
    (void)in_sizes; (void)n_in; (void)out_size; (void)ws_size;
    const float* x     = (const float*)d_in[0];
    const float* ln1_g = (const float*)d_in[1];
    const float* ln1_b = (const float*)d_in[2];
    const float* w_qkv = (const float*)d_in[3];
    const float* w_out = (const float*)d_in[4];
    const float* b_out = (const float*)d_in[5];
    const float* ln2_g = (const float*)d_in[6];
    const float* ln2_b = (const float*)d_in[7];
    const float* w1    = (const float*)d_in[8];
    const float* b1    = (const float*)d_in[9];
    const float* w2    = (const float*)d_in[10];
    const float* b2    = (const float*)d_in[11];
    float* h = (float*)d_out;   // residual stream lives in d_out

    const size_t WQKV = 6ull * DIM * QKV3;
    const size_t WOUT = 6ull * DIM * DIM;
    const size_t W1E  = 6ull * DIM * MLP_D;
    const size_t W2E  = 6ull * MLP_D * DIM;

    unsigned short* wsp = (unsigned short*)d_ws;
    size_t off = 0;
    auto alloc = [&](size_t elems) {
        unsigned short* p = wsp + off;
        off += (elems + 127) & ~(size_t)127;
        return p;
    };
    unsigned short* wqkv_bf = alloc(WQKV);   // [3072][1024] per depth (transposed)
    unsigned short* wout_bf = alloc(WOUT);   // [1024][1024] per depth (transposed)
    unsigned short* w1_bf   = alloc(W1E);    // [4096][1024] per depth (transposed)
    unsigned short* w2_bf   = alloc(W2E);    // [1024][4096] per depth (transposed)
    unsigned short* xn      = alloc((size_t)ROWS * DIM);
    unsigned short* qkvb    = alloc((size_t)ROWS * QKV3);
    unsigned short* ob      = alloc((size_t)ROWS * DIM);
    unsigned short* fb      = alloc((size_t)ROWS * MLP_D);

    // residual = x
    hipMemcpyAsync(h, x, (size_t)ROWS * DIM * sizeof(float),
                   hipMemcpyDeviceToDevice, stream);

    // per-launch weight convert + transpose to bf16 [N][K]
    transpose_f32_to_bf16<<<dim3(QKV3 / 32, DIM / 32, 6), 256, 0, stream>>>(
        w_qkv, wqkv_bf, DIM, QKV3);
    transpose_f32_to_bf16<<<dim3(DIM / 32, DIM / 32, 6), 256, 0, stream>>>(
        w_out, wout_bf, DIM, DIM);
    transpose_f32_to_bf16<<<dim3(MLP_D / 32, DIM / 32, 6), 256, 0, stream>>>(
        w1, w1_bf, DIM, MLP_D);
    transpose_f32_to_bf16<<<dim3(DIM / 32, MLP_D / 32, 6), 256, 0, stream>>>(
        w2, w2_bf, MLP_D, DIM);

    for (int l = 0; l < 6; ++l) {
        const size_t oq = (size_t)l * DIM * QKV3;
        const size_t oo = (size_t)l * DIM * DIM;
        const size_t o1 = (size_t)l * DIM * MLP_D;
        const size_t o2 = (size_t)l * MLP_D * DIM;
        const size_t d1 = (size_t)l * DIM;
        const size_t dm = (size_t)l * MLP_D;

        // --- attention block ---
        layernorm_to_bf16<<<ROWS, 256, 0, stream>>>(h, ln1_g + d1, ln1_b + d1, xn);
        gemm_bf16_wmma<0, 0, 0, 1><<<dim3(QKV3 / BN, ROWS / BM), 256, 0, stream>>>(
            xn, wqkv_bf + oq, nullptr, nullptr, qkvb, ROWS, QKV3, DIM);
        attention_wmma<<<dim3(SEQ / 64, HEADS, BATCH), 128, 0, stream>>>(qkvb, ob);
        gemm_bf16_wmma<1, 0, 1, 0><<<dim3(DIM / BN, ROWS / BM), 256, 0, stream>>>(
            ob, wout_bf + oo, b_out + d1, h, nullptr, ROWS, DIM, DIM);

        // --- MLP block ---
        layernorm_to_bf16<<<ROWS, 256, 0, stream>>>(h, ln2_g + d1, ln2_b + d1, xn);
        gemm_bf16_wmma<1, 1, 0, 1><<<dim3(MLP_D / BN, ROWS / BM), 256, 0, stream>>>(
            xn, w1_bf + o1, b1 + dm, nullptr, fb, ROWS, MLP_D, DIM);
        gemm_bf16_wmma<1, 0, 1, 0><<<dim3(DIM / BN, ROWS / BM), 256, 0, stream>>>(
            fb, w2_bf + o2, b2 + d1, h, nullptr, ROWS, DIM, MLP_D);
    }
}
